// TSF_44461501448858
// MI455X (gfx1250) — matrix operand order repaired and checked by reading, not compiled, not run on hardware
//
#include <hip/hip_runtime.h>
#include <cstdint>
#include <cstddef>

// CDNA5 / gfx1250 wave32 WMMA + TDM types
typedef __attribute__((ext_vector_type(16))) _Float16     v16h;
typedef __attribute__((ext_vector_type(8)))  float        v8f;
typedef __attribute__((ext_vector_type(4)))  unsigned int v4u;
typedef __attribute__((ext_vector_type(8)))  int          v8i;
typedef __attribute__((ext_vector_type(4)))  int          v4i;

#define C_IN   1024
#define C_OUT  32
#define T_DIM  64
#define W_DIM  14
#define H_DIM  14
#define SPAT   (T_DIM * W_DIM * H_DIM)   // 12544
#define B_DIM  8
#define NI     3
#define NTILE  64                         // spatial columns per workgroup
#define KTILE  32                         // K step per WMMA
#define NKT    (C_IN / KTILE)             // 32 K-steps
#define POOLED 9216                       // 32*32*3*3
#define FC1N   1024

#if __has_builtin(__builtin_amdgcn_tensor_load_to_lds) && __has_builtin(__builtin_amdgcn_s_wait_tensorcnt)
#define HAVE_TDM 1
#else
#define HAVE_TDM 0
#endif

// ---------------------------------------------------------------------------
// K1: pre-swizzle conv_w (32 x 1024 f32) into WMMA A-fragment layout, f16.
// A-matrix 16x32 f16 layout (ISA 7.12.2): lane L holds M = L&15;
// half i (0..15): K = (i>>3)*16 + ((L>=16)?8:0) + (i&7).
// aFrag[ktile][mtile][lane][16] halves; ktile=K/32, mtile=M/16.  64 KB blob.
// ---------------------------------------------------------------------------
__global__ void k_prep_a(const float* __restrict__ conv_w,
                         _Float16* __restrict__ aFrag) {
    int tid = blockIdx.x * blockDim.x + threadIdx.x;   // 0..32767
    if (tid >= NKT * 2 * 32 * 16) return;
    int i     = tid & 15;
    int lane  = (tid >> 4) & 31;
    int mtile = (tid >> 9) & 1;
    int ktile = tid >> 10;
    int m = mtile * 16 + (lane & 15);
    int k = ktile * 32 + (i >> 3) * 16 + ((lane >> 4) & 1) * 8 + (i & 7);
    aFrag[tid] = (_Float16)conv_w[m * C_IN + k];
}

#if HAVE_TDM
// Issue a TDM load of one contiguous 2 KB A-fragment slab into LDS.
// D# per ISA 8.3/8.4: group0 = {count=1, lds_addr, global_addr, type=2};
// group1 = {data_size=4B, tensor_dim0=tile_dim0=512 dwords, dim1=1}.
// 6-arg builtin form: (v4u g0, v8i g1, v4i g2, v4i g3, v8i g4, i32 cpol).
__device__ __forceinline__ void tdm_load_2k(const _Float16* gsrc, unsigned lds_off) {
    unsigned long long ga = (unsigned long long)(size_t)gsrc;
    v4u g0 = {};
    g0[0] = 1u;                                             // count=1 (valid user D#)
    g0[1] = lds_off;                                        // LDS byte address
    g0[2] = (unsigned)ga;                                   // global_addr[31:0]
    g0[3] = (unsigned)((ga >> 32) & 0x01FFFFFFu) | (2u << 30); // addr[56:32] | type=2
    const unsigned ND = (2 * 32 * 16 * 2) / 4;              // 512 dwords = 2 KB
    v8i g1 = {};
    g1[0] = (int)(2u << 16);                                // wgmask=0, data_size=2 (4B)
    g1[1] = (int)(ND << 16);                                // tensor_dim0[15:0]
    g1[2] = (int)(1u << 16);                                // tensor_dim0[31:16]=0, tensor_dim1=1
    g1[3] = (int)(ND << 16);                                // tile_dim0 = 512
    g1[4] = 0;                                              // tile_dim1/2 unused
    g1[5] = (int)ND;                                        // tensor_dim0_stride
    g1[6] = 0; g1[7] = 0;
    v4i z4 = {};
    v8i z8 = {};
    __builtin_amdgcn_tensor_load_to_lds(g0, g1, z4, z4, z8, 0);
}
#endif

// Stage one 32(K) x 64(n) f32 video slab -> f16 LDS in B-fragment order.
// B-matrix 32x16 f16 layout: lane L holds N=L&15; half i: K=(i>>3)*16 + ((L>=16)?8:0) + (i&7).
// buf[(nsub*32 + lane)*16 + half].  float4 (b128) global reads.
__device__ __forceinline__ void stage_B(const float* __restrict__ vsrc,
                                        _Float16* buf, int tid) {
    #pragma unroll
    for (int it = 0; it < 2; ++it) {
        int e4 = tid + 256 * it;                 // 0..511
        int k  = e4 >> 4;                        // 0..31
        int n0 = (e4 & 15) * 4;                  // 0..60
        const float4 v = *(const float4*)(vsrc + (size_t)k * SPAT + n0);
        int klo = (k & 7) | ((k & 16) >> 1);
        int khi = (k & 8) ? 16 : 0;
        float vv[4] = {v.x, v.y, v.z, v.w};
        #pragma unroll
        for (int q = 0; q < 4; ++q) {
            int n = n0 + q;
            int dst = (((n >> 4) * 32) + (n & 15) + khi) * 16 + klo;
            buf[dst] = (_Float16)vv[q];
        }
    }
}

// ---------------------------------------------------------------------------
// K2: 1x1 conv via v_wmma_f32_16x16x32_f16 + leaky_relu.
// Per workgroup: one b, 64 spatial cols. 8 waves = 2 Mtiles x 4 Ntiles.
// Double-buffered LDS; A slabs arrive via TDM (tensor_load_to_lds).
// ---------------------------------------------------------------------------
__global__ void __launch_bounds__(256)
k_conv_wmma(const float* __restrict__ video,
            const _Float16* __restrict__ aFrag,
            const float* __restrict__ conv_b,
            float* __restrict__ attn) {
#if HAVE_TDM
    __shared__ _Float16 ldsA[2][2 * 32 * 16];  // 2 x 2 KB A slabs
#endif
    __shared__ _Float16 ldsB[2][4 * 32 * 16];  // 2 x 4 KB B slabs
    const int b     = blockIdx.y;
    const int ntile = blockIdx.x * NTILE;
    const int tid   = threadIdx.x;
    const int lane  = tid & 31;
    const int wave  = tid >> 5;
    const int mtile = wave & 1;
    const int nsub  = wave >> 1;
    const float* vbase = video + (size_t)b * C_IN * SPAT + ntile;

    // Prologue: stage K-step 0.
    stage_B(vbase, ldsB[0], tid);
#if HAVE_TDM
    if (tid == 0) {                                  // wave 0 issues TDM (EXEC ignored)
        tdm_load_2k(aFrag, (unsigned)(size_t)&ldsA[0][0]);
        __builtin_amdgcn_s_wait_tensorcnt(0);
    }
#endif
    __syncthreads();

    v8f acc = {};
    for (int kt = 0; kt < NKT; ++kt) {
        const int p = kt & 1;
        if (kt + 1 < NKT) {                          // prefetch next slab into other buffer
            stage_B(vbase + (size_t)(kt + 1) * KTILE * SPAT, ldsB[1 - p], tid);
#if HAVE_TDM
            if (tid == 0)
                tdm_load_2k(aFrag + (size_t)(kt + 1) * (2 * 32 * 16),
                            (unsigned)(size_t)&ldsA[1 - p][0]);
#endif
        }
#if HAVE_TDM
        v16h a = *(const v16h*)(&ldsA[p][(mtile * 32 + lane) * 16]);
#else
        v16h a = *(const v16h*)(aFrag + (size_t)((kt * 2 + mtile) * 32 + lane) * 16);
#endif
        v16h bf = *(const v16h*)(&ldsB[p][(nsub * 32 + lane) * 16]);
        // (neg_a, A, neg_b, B, c_mod, C, reuse_a, reuse_b)
        acc = __builtin_amdgcn_wmma_f32_16x16x32_f16(false, a, false, bf,
                                                     (short)0, acc, false, false);
#if HAVE_TDM
        if (tid == 0) __builtin_amdgcn_s_wait_tensorcnt(0);
#endif
        __syncthreads();
    }

    // D layout: VGPR r of lane L -> N = L&15, M = r + ((L>=16)?8:0).
    const int ncol  = ntile + nsub * 16 + (lane & 15);
    const int mbase = mtile * 16 + ((lane >= 16) ? 8 : 0);
    #pragma unroll
    for (int r = 0; r < 8; ++r) {
        int m = mbase + r;
        float x = acc[r] + conv_b[m];
        x = (x >= 0.f) ? x : 0.01f * x;                 // leaky_relu 0.01
        attn[(size_t)(b * C_OUT + m) * SPAT + ncol] = x;
    }
}

// ---------------------------------------------------------------------------
// K3a: adaptive pool (64,14,14) -> (32,3,3). Overlapping bins per JAX rule.
// ---------------------------------------------------------------------------
__global__ void k_pool(const float* __restrict__ attn, float* __restrict__ pooled) {
    int b = blockIdx.x;
    for (int p = threadIdx.x; p < POOLED; p += blockDim.x) {
        int j = p % 3, i = (p / 3) % 3, a = (p / 9) % 32, o = p / 288;
        int ts = a * 2,            te = a * 2 + 2;
        int ws0 = (i * 14) / 3,    we = ((i + 1) * 14 + 2) / 3;
        int hs = (j * 14) / 3,     he = ((j + 1) * 14 + 2) / 3;
        const float* base = attn + (size_t)(b * C_OUT + o) * SPAT;
        float s = 0.f; int cnt = 0;
        for (int t = ts; t < te; ++t)
            for (int w = ws0; w < we; ++w)
                for (int h = hs; h < he; ++h) { s += base[t * 196 + w * 14 + h]; ++cnt; }
        pooled[b * POOLED + p] = s / (float)cnt;
    }
}

// ---------------------------------------------------------------------------
// K3b: FC1 (9216 -> 1024) + leaky_relu. One wave per output feature;
// each fc11_w row (37 KB, b128 loads) is read ONCE, reused for all 8 batches.
// ---------------------------------------------------------------------------
__global__ void k_fc1(const float* __restrict__ fc11_w,
                      const float* __restrict__ fc11_b,
                      const float* __restrict__ pooled,
                      float* __restrict__ a1) {
    int gw   = blockIdx.x * (blockDim.x >> 5) + (threadIdx.x >> 5);
    int lane = threadIdx.x & 31;
    if (gw >= FC1N) return;
    float part[B_DIM] = {};
    const float4* wrow = (const float4*)(fc11_w + (size_t)gw * POOLED);
    const float4* pp   = (const float4*)pooled;
    for (int k4 = lane; k4 < POOLED / 4; k4 += 32) {
        float4 wv = wrow[k4];
        #pragma unroll
        for (int b = 0; b < B_DIM; ++b) {
            float4 pv = pp[b * (POOLED / 4) + k4];
            part[b] += wv.x * pv.x + wv.y * pv.y + wv.z * pv.z + wv.w * pv.w;
        }
    }
    #pragma unroll
    for (int b = 0; b < B_DIM; ++b) {
        float s = part[b];
        for (int off = 16; off > 0; off >>= 1) s += __shfl_down(s, off, 32);
        if (lane == 0) {
            float x = s + fc11_b[gw];
            a1[b * FC1N + gw] = (x >= 0.f) ? x : 0.01f * x;
        }
    }
}

// ---------------------------------------------------------------------------
// K3c: FC2 + sigmoid -> mu; build normalized Gaussian map f[b,n,t,w,h].
// One workgroup per batch.
// ---------------------------------------------------------------------------
__global__ void k_attnmap(const float* __restrict__ a1,
                          const float* __restrict__ fc12_w,
                          const float* __restrict__ fc12_b,
                          const int* __restrict__ length,
                          float* __restrict__ f) {
    __shared__ float red[256];
    __shared__ float z[9];
    __shared__ float mu_t[3], mu_x[3], mu_y[3], iv_t[3], iv_x[3], iv_y[3], nsum[3];
    int b = blockIdx.x, tid = threadIdx.x;

    for (int r = 0; r < 9; ++r) {
        float p = 0.f;
        for (int k = tid; k < FC1N; k += 256) p += fc12_w[r * FC1N + k] * a1[b * FC1N + k];
        red[tid] = p; __syncthreads();
        for (int s = 128; s > 0; s >>= 1) { if (tid < s) red[tid] += red[tid + s]; __syncthreads(); }
        if (tid == 0) {
            float x = 0.001f * (red[0] + fc12_b[r]);
            z[r] = 1.f / (1.f + expf(-x));
        }
        __syncthreads();
    }
    if (tid == 0) {
        const float SIG[3] = {-0.5f, 0.f, 0.5f};
        float len_f = (float)length[b];
        for (int n = 0; n < 3; ++n) {
            float st = expf(7.f * fabsf(SIG[n]) - 1.5f);
            float sx = expf(5.f * fabsf(SIG[n]) - 1.5f);
            iv_t[n] = 1.f / (st * st + 1e-6f);
            iv_x[n] = 1.f / (sx * sx + 1e-6f);
            iv_y[n] = iv_x[n];
            mu_t[n] = (len_f - 1.f) * (z[n] + 1.f) * 0.5f;
            mu_x[n] = (float)(W_DIM - 1) * (z[3 + n] + 1.f) * 0.5f;
            mu_y[n] = (float)(H_DIM - 1) * (z[6 + n] + 1.f) * 0.5f;
        }
    }
    __syncthreads();

    float loc[3] = {0.f, 0.f, 0.f};
    float* fb = f + (size_t)b * NI * SPAT;
    for (int e = tid; e < NI * SPAT; e += 256) {
        int n = e / SPAT;
        int rem = e - n * SPAT;
        int t = rem / 196;
        int r2 = rem - t * 196;
        int w = r2 / 14;
        int h = r2 - w * 14;
        float dt = (float)t - mu_t[n], dx = (float)w - mu_x[n], dy = (float)h - mu_y[n];
        float expo = dt * dt * iv_t[n] + dx * dx * iv_x[n] + dy * dy * iv_y[n];
        float val = expf(-0.5f * expo);
        fb[e] = val;
        loc[n] += val;
    }
    for (int n = 0; n < 3; ++n) {
        red[tid] = loc[n]; __syncthreads();
        for (int s = 128; s > 0; s >>= 1) { if (tid < s) red[tid] += red[tid + s]; __syncthreads(); }
        if (tid == 0) nsum[n] = red[0];
        __syncthreads();
    }
    for (int e = tid; e < NI * SPAT; e += 256) {
        int n = e / SPAT;
        fb[e] = fb[e] / (nsum[n] + 1e-6f);
    }
}

// ---------------------------------------------------------------------------
// K4: o[b,c,n] = sum_thw f[b,n,thw] * video[b,c,thw]. Bandwidth-bound pass
// (411 MB of video); one wave per (b,c), lanes stride spatial via b128 loads.
// ---------------------------------------------------------------------------
__global__ void __launch_bounds__(256)
k_out(const float* __restrict__ video,
      const float* __restrict__ f,
      float* __restrict__ out) {
    int gw   = blockIdx.x * (blockDim.x >> 5) + (threadIdx.x >> 5);  // 0..8191
    int lane = threadIdx.x & 31;
    int b = gw >> 10;
    int c = gw & 1023;
    const float* vb = video + (size_t)(b * C_IN + c) * SPAT;
    const float* fb = f + (size_t)b * NI * SPAT;
    float s0 = 0.f, s1 = 0.f, s2 = 0.f;
    for (int i = lane * 4; i < SPAT; i += 128) {     // SPAT = 98 * 128, exact
        if (i + 2048 < SPAT)
            __builtin_prefetch(vb + i + 2048, 0, 0); // global_prefetch_b8
        float4 v  = *(const float4*)(vb + i);
        float4 f0 = *(const float4*)(fb + i);
        float4 f1 = *(const float4*)(fb + SPAT + i);
        float4 f2 = *(const float4*)(fb + 2 * SPAT + i);
        s0 += v.x * f0.x + v.y * f0.y + v.z * f0.z + v.w * f0.w;
        s1 += v.x * f1.x + v.y * f1.y + v.z * f1.z + v.w * f1.w;
        s2 += v.x * f2.x + v.y * f2.y + v.z * f2.z + v.w * f2.w;
    }
    for (int off = 16; off > 0; off >>= 1) {
        s0 += __shfl_down(s0, off, 32);
        s1 += __shfl_down(s1, off, 32);
        s2 += __shfl_down(s2, off, 32);
    }
    if (lane == 0) {
        out[b * (C_IN * NI) + c * NI + 0] = s0;
        out[b * (C_IN * NI) + c * NI + 1] = s1;
        out[b * (C_IN * NI) + c * NI + 2] = s2;
    }
}

// ---------------------------------------------------------------------------
extern "C" void kernel_launch(void* const* d_in, const int* in_sizes, int n_in,
                              void* d_out, int out_size, void* d_ws, size_t ws_size,
                              hipStream_t stream) {
    const float* video  = (const float*)d_in[0];
    const int*   length = (const int*)  d_in[1];
    const float* conv_w = (const float*)d_in[2];
    const float* conv_b = (const float*)d_in[3];
    const float* fc11_w = (const float*)d_in[4];
    const float* fc11_b = (const float*)d_in[5];
    const float* fc12_w = (const float*)d_in[6];
    const float* fc12_b = (const float*)d_in[7];
    float* out = (float*)d_out;

    // Workspace layout (~13.8 MB total)
    char* ws = (char*)d_ws;
    _Float16* aFrag = (_Float16*)ws;                               // 64 KB
    size_t off = 64 * 1024;
    float* attn = (float*)(ws + off);                              // 12.25 MB
    off += (size_t)B_DIM * C_OUT * SPAT * sizeof(float);
    float* pooled = (float*)(ws + off);                            // 288 KB
    off += (size_t)B_DIM * POOLED * sizeof(float);
    float* a1 = (float*)(ws + off);                                // 32 KB
    off += (size_t)B_DIM * FC1N * sizeof(float);
    float* f = (float*)(ws + off);                                 // 1.15 MB

    k_prep_a   <<<128, 256, 0, stream>>>(conv_w, aFrag);
    k_conv_wmma<<<dim3(SPAT / NTILE, B_DIM), 256, 0, stream>>>(video, aFrag, conv_b, attn);
    k_pool     <<<B_DIM, 256, 0, stream>>>(attn, pooled);
    k_fc1      <<<128, 256, 0, stream>>>(fc11_w, fc11_b, pooled, a1);
    k_attnmap  <<<B_DIM, 256, 0, stream>>>(a1, fc12_w, fc12_b, length, f);
    k_out      <<<1024, 256, 0, stream>>>(video, f, out);
}
